// CoLES_78451872628886
// MI455X (gfx1250) — matrix operand order repaired and checked by reading, not compiled, not run on hardware
//
#include <hip/hip_runtime.h>
#include <stdint.h>

// ---------------- types / constants ----------------
typedef __attribute__((ext_vector_type(16))) __bf16 v16bf;
typedef __attribute__((ext_vector_type(8)))  float  v8f;
typedef __attribute__((ext_vector_type(4)))  float  f32x4;

#define Bsz  512
#define Ssz  512
#define Vsz  1024
#define Hsz  512
#define Esz  256
#define Lsz  64
#define Ksub 2
#define NSEQ 1024        // Ksub*Bsz
#define G3H  1536        // 3*Hsz
#define TAUI 10.0f       // 1/TAU

__device__ __forceinline__ unsigned short f2bf(float f) {
  unsigned u = __float_as_uint(f);
  u += 0x7FFFu + ((u >> 16) & 1u);      // round-to-nearest-even
  return (unsigned short)(u >> 16);
}
__device__ __forceinline__ float sigm(float x) { return 1.0f / (1.0f + __expf(-x)); }

// ---------------- prep kernels ----------------
// W_ihT[v][g] = W_ih[g][v]   (fp32, one-hot einsum becomes a row gather)
__global__ void k_transpose_wih(const float* __restrict__ W_ih, float* __restrict__ W_ihT) {
  int idx = blockIdx.x * blockDim.x + threadIdx.x;
  if (idx >= Vsz * G3H) return;
  int v = idx / G3H, g = idx % G3H;
  W_ihT[idx] = W_ih[g * Vsz + v];
}
// Bmat[k][g] = bf16(W_hh[g][k])  -> B operand of gh GEMM (K x N row-major)
__global__ void k_prep_whh(const float* __restrict__ W_hh, unsigned short* __restrict__ Bmat) {
  int idx = blockIdx.x * blockDim.x + threadIdx.x;
  if (idx >= Hsz * G3H) return;
  int k = idx / G3H, g = idx % G3H;
  Bmat[idx] = f2bf(W_hh[g * Hsz + k]);
}
// tok[n][l] = sequence[b, starts[k,b]+l]
__global__ void k_tokens(const int* __restrict__ seq, const int* __restrict__ starts,
                         int* __restrict__ tok) {
  int idx = blockIdx.x * blockDim.x + threadIdx.x;
  if (idx >= NSEQ * Lsz) return;
  int n = idx / Lsz, l = idx % Lsz;
  int k = n / Bsz, b = n % Bsz;
  int st = starts[k * Bsz + b];
  tok[idx] = seq[b * Ssz + st + l];
}
__global__ void k_init(unsigned short* __restrict__ h_bf, float* __restrict__ h_f,
                       float* __restrict__ pooled) {
  int idx = blockIdx.x * blockDim.x + threadIdx.x;
  if (idx >= NSEQ * Hsz) return;
  h_bf[idx] = 0; h_f[idx] = 0.0f; pooled[idx] = 0.0f;
}

// ---------------- bf16 WMMA GEMM: C[M,N] = alpha * A[M,K] * B[K,N] ----------------
// A,B row-major bf16 (as u16), C fp32. Block = 128 thr (4 waves), tile 64x64, BK=32.
#define BM 64
#define BN 64
#define BK 32
#define LDT 48   // padded LDS row (elems); 96B rows keep 16B-aligned frag reads
__global__ __launch_bounds__(128) void k_gemm_bf16(
    const unsigned short* __restrict__ A, const unsigned short* __restrict__ Bm,
    float* __restrict__ C, int M, int N, int K, float alpha) {
  __shared__ __align__(16) unsigned short As[BM * LDT];   // [m][k]
  __shared__ __align__(16) unsigned short Bs[BN * LDT];   // [n][k] (transposed stage)
  const int tid = threadIdx.x;
  const int lane = tid & 31, wave = tid >> 5;
  const int m0 = blockIdx.y * BM, n0 = blockIdx.x * BN;
  const int wr = (wave >> 1) * 32;   // wave row offset in tile
  const int wc = (wave & 1) * 32;    // wave col offset in tile

  v8f acc[2][2];
#pragma unroll
  for (int i = 0; i < 2; ++i)
#pragma unroll
    for (int j = 0; j < 2; ++j)
#pragma unroll
      for (int r = 0; r < 8; ++r) acc[i][j][r] = 0.0f;

  union Frag { v16bf v; f32x4 f4[2]; };
  union Pack { f32x4 d; unsigned short u[8]; };

  for (int kc = 0; kc < K; kc += BK) {
    // ---- stage A tile (64x32): 2 x (8 bf16 = 16B) per thread
#pragma unroll
    for (int i = 0; i < 2; ++i) {
      int s = tid + i * 128;                 // 0..255
      int row = s >> 2, seg = (s & 3) * 8;
      f32x4 d = *(const f32x4*)(A + (size_t)(m0 + row) * K + kc + seg);
      *(f32x4*)(&As[row * LDT + seg]) = d;
    }
    // ---- stage B tile transposed (Bs[n][k]): read 8 contiguous n at fixed k, scatter
#pragma unroll
    for (int i = 0; i < 2; ++i) {
      int s = tid + i * 128;
      int kk = s >> 3, nseg = (s & 7) * 8;
      Pack p; p.d = *(const f32x4*)(Bm + (size_t)(kc + kk) * N + n0 + nseg);
#pragma unroll
      for (int e = 0; e < 8; ++e) Bs[(nseg + e) * LDT + kk] = p.u[e];
    }
    __syncthreads();

    // ---- fragment loads per ISA 7.12.2 layouts
    const int klo = (lane < 16) ? 0 : 8;     // A: K halves per lane group
    const int ks  = (lane < 16) ? 0 : 8;     // B: K halves per lane group
    Frag a[2], b[2];
#pragma unroll
    for (int mi = 0; mi < 2; ++mi) {
      int m = wr + mi * 16 + (lane & 15);
      a[mi].f4[0] = *(const f32x4*)(&As[m * LDT + klo]);        // K = klo..klo+7
      a[mi].f4[1] = *(const f32x4*)(&As[m * LDT + klo + 16]);   // K = klo+16..klo+23
    }
#pragma unroll
    for (int ni = 0; ni < 2; ++ni) {
      int n = wc + ni * 16 + (lane & 15);
      b[ni].f4[0] = *(const f32x4*)(&Bs[n * LDT + ks]);         // K = ks..ks+7
      b[ni].f4[1] = *(const f32x4*)(&Bs[n * LDT + ks + 16]);    // K = ks+16..ks+23
    }
#pragma unroll
    for (int mi = 0; mi < 2; ++mi)
#pragma unroll
      for (int ni = 0; ni < 2; ++ni)
        acc[mi][ni] = __builtin_amdgcn_wmma_f32_16x16x32_bf16(
            false, a[mi].v, false, b[ni].v, (short)0, acc[mi][ni], false, false);
    __syncthreads();
  }

  // ---- store D: lane -> n = lane&15 (+16 lanes carry m+8), vgpr r -> m = r (+8)
  const int nlane = lane & 15;
  const int mo = (lane >= 16) ? 8 : 0;
#pragma unroll
  for (int mi = 0; mi < 2; ++mi)
#pragma unroll
    for (int ni = 0; ni < 2; ++ni)
#pragma unroll
      for (int r = 0; r < 8; ++r) {
        int m = m0 + wr + mi * 16 + mo + r;
        int n = n0 + wc + ni * 16 + nlane;
        C[(size_t)m * N + n] = alpha * acc[mi][ni][r];
      }
}

// ---------------- fused GRU gate update + pooled accumulation ----------------
__global__ __launch_bounds__(256) void k_gate(
    const float* __restrict__ gh, const float* __restrict__ W_ihT,
    const float* __restrict__ b_ih, const float* __restrict__ b_hh,
    const int* __restrict__ tok, int l,
    unsigned short* __restrict__ h_bf, float* __restrict__ h_f,
    float* __restrict__ pooled) {
  int idx = blockIdx.x * blockDim.x + threadIdx.x;
  if (idx >= NSEQ * Hsz) return;
  int n = idx / Hsz, j = idx % Hsz;
  const float* gx = W_ihT + (size_t)tok[n * Lsz + l] * G3H;   // one-hot einsum = row gather
  float xr = gx[j]            + b_ih[j];
  float xz = gx[Hsz + j]      + b_ih[Hsz + j];
  float xn = gx[2 * Hsz + j]  + b_ih[2 * Hsz + j];
  const float* ghr = gh + (size_t)n * G3H;
  float hr = ghr[j]           + b_hh[j];
  float hz = ghr[Hsz + j]     + b_hh[Hsz + j];
  float hn = ghr[2 * Hsz + j] + b_hh[2 * Hsz + j];
  float r = sigm(xr + hr);
  float z = sigm(xz + hz);
  float nn = tanhf(xn + r * hn);
  float hprev = h_f[idx];
  float hnew = (1.0f - z) * nn + z * hprev;
  h_f[idx] = hnew;
  h_bf[idx] = f2bf(hnew);
  pooled[idx] += hnew;
}

// ---------------- projection: z = (pooled/L) @ proj_W^T + proj_b ----------------
__global__ __launch_bounds__(256) void k_proj(
    const float* __restrict__ pooled, const float* __restrict__ proj_W,
    const float* __restrict__ proj_b, float* __restrict__ z) {
  int idx = blockIdx.x * blockDim.x + threadIdx.x;
  if (idx >= NSEQ * Esz) return;
  int n = idx / Esz, e = idx % Esz;
  const float* pr = pooled + (size_t)n * Hsz;
  const float* wr = proj_W + (size_t)e * Hsz;
  float s = 0.0f;
#pragma unroll 4
  for (int j = 0; j < Hsz; ++j) s += pr[j] * wr[j];
  z[idx] = s * (1.0f / (float)Lsz) + proj_b[e];
}

// ---------------- row-normalize, emit zn (bf16) and znT (bf16) ----------------
__global__ __launch_bounds__(256) void k_norm(
    const float* __restrict__ z, unsigned short* __restrict__ zn,
    unsigned short* __restrict__ znT) {
  __shared__ float red[256];
  int n = blockIdx.x, t = threadIdx.x;
  float v = z[(size_t)n * Esz + t];
  red[t] = v * v;
  __syncthreads();
  for (int s = 128; s > 0; s >>= 1) { if (t < s) red[t] += red[t + s]; __syncthreads(); }
  float nrm = fmaxf(sqrtf(red[0]), 1e-12f);
  float zv = v / nrm;
  unsigned short b = f2bf(zv);
  zn[(size_t)n * Esz + t] = b;
  znT[(size_t)t * NSEQ + n] = b;
}

// ---------------- SupCon loss per row ----------------
__global__ __launch_bounds__(256) void k_loss_rows(
    const float* __restrict__ sim, const int* __restrict__ labels,
    float* __restrict__ li_arr, int* __restrict__ hp_arr) {
  __shared__ float rnum[256], rden[256];
  __shared__ int rpos[256];
  int i = blockIdx.x, t = threadIdx.x;
  int lab_i = labels[i % Bsz];
  float num = 0.0f, den = 0.0f; int pc = 0;
  for (int j = t; j < NSEQ; j += 256) {
    if (j == i) continue;
    float e = __expf(sim[(size_t)i * NSEQ + j]);
    bool p = (labels[j % Bsz] == lab_i);
    num += p ? e : 0.0f; den += e; pc += p ? 1 : 0;
  }
  rnum[t] = num; rden[t] = den; rpos[t] = pc;
  __syncthreads();
  for (int s = 128; s > 0; s >>= 1) {
    if (t < s) { rnum[t] += rnum[t + s]; rden[t] += rden[t + s]; rpos[t] += rpos[t + s]; }
    __syncthreads();
  }
  if (t == 0) {
    bool hp = rpos[0] > 0;
    float li = -logf(rnum[0] / (rden[0] + 1e-10f) + 1e-10f);
    li_arr[i] = hp ? li : 0.0f;
    hp_arr[i] = hp ? 1 : 0;
  }
}

__global__ __launch_bounds__(256) void k_final(
    const float* __restrict__ li_arr, const int* __restrict__ hp_arr,
    float* __restrict__ out) {
  __shared__ float rs[256];
  __shared__ int rc[256];
  int t = threadIdx.x;
  float s = 0.0f; int c = 0;
  for (int i = t; i < NSEQ; i += 256) { s += li_arr[i]; c += hp_arr[i]; }
  rs[t] = s; rc[t] = c;
  __syncthreads();
  for (int k = 128; k > 0; k >>= 1) {
    if (t < k) { rs[t] += rs[t + k]; rc[t] += rc[t + k]; }
    __syncthreads();
  }
  if (t == 0) out[0] = rs[0] / (float)max(rc[0], 1);
}

// ---------------- launch ----------------
extern "C" void kernel_launch(void* const* d_in, const int* in_sizes, int n_in,
                              void* d_out, int out_size, void* d_ws, size_t ws_size,
                              hipStream_t stream) {
  const int*   seq    = (const int*)d_in[0];
  const int*   labels = (const int*)d_in[1];
  const int*   starts = (const int*)d_in[2];
  const float* W_ih   = (const float*)d_in[3];
  const float* W_hh   = (const float*)d_in[4];
  const float* b_ih   = (const float*)d_in[5];
  const float* b_hh   = (const float*)d_in[6];
  const float* proj_W = (const float*)d_in[7];
  const float* proj_b = (const float*)d_in[8];
  float* out = (float*)d_out;

  char* ws = (char*)d_ws;
  size_t o = 0;
  auto carve = [&](size_t bytes) { char* p = ws + o; o = (o + bytes + 255) & ~(size_t)255; return p; };
  float*          W_ihT  = (float*)         carve((size_t)Vsz * G3H * 4);   // 6 MB
  unsigned short* Bmat   = (unsigned short*)carve((size_t)Hsz * G3H * 2);   // 1.5 MB
  int*            tok    = (int*)           carve((size_t)NSEQ * Lsz * 4);  // 256 KB
  unsigned short* h_bf   = (unsigned short*)carve((size_t)NSEQ * Hsz * 2);  // 1 MB
  float*          h_f    = (float*)         carve((size_t)NSEQ * Hsz * 4);  // 2 MB
  float*          pooled = (float*)         carve((size_t)NSEQ * Hsz * 4);  // 2 MB
  float*          gh     = (float*)         carve((size_t)NSEQ * G3H * 4);  // 6 MB
  float*          zbuf   = (float*)         carve((size_t)NSEQ * Esz * 4);  // 1 MB
  unsigned short* zn     = (unsigned short*)carve((size_t)NSEQ * Esz * 2);  // 0.5 MB
  unsigned short* znT    = (unsigned short*)carve((size_t)Esz * NSEQ * 2);  // 0.5 MB
  float*          sim    = (float*)         carve((size_t)NSEQ * NSEQ * 4); // 4 MB
  float*          li_arr = (float*)         carve((size_t)NSEQ * 4);
  int*            hp_arr = (int*)           carve((size_t)NSEQ * 4);
  (void)ws_size; (void)n_in; (void)in_sizes; (void)out_size;

  // prep
  k_transpose_wih<<<(Vsz * G3H + 255) / 256, 256, 0, stream>>>(W_ih, W_ihT);
  k_prep_whh<<<(Hsz * G3H + 255) / 256, 256, 0, stream>>>(W_hh, Bmat);
  k_tokens<<<(NSEQ * Lsz + 255) / 256, 256, 0, stream>>>(seq, starts, tok);
  k_init<<<(NSEQ * Hsz + 255) / 256, 256, 0, stream>>>(h_bf, h_f, pooled);

  // GRU recurrence: 64 x (WMMA GEMM 1024x1536x512 + fused gates)
  dim3 grid_gh(G3H / BN, NSEQ / BM);   // 24 x 16 = 384 WGs
  for (int l = 0; l < Lsz; ++l) {
    k_gemm_bf16<<<grid_gh, 128, 0, stream>>>(h_bf, Bmat, gh, NSEQ, G3H, Hsz, 1.0f);
    k_gate<<<(NSEQ * Hsz + 255) / 256, 256, 0, stream>>>(
        gh, W_ihT, b_ih, b_hh, tok, l, h_bf, h_f, pooled);
  }

  // projection + normalize
  k_proj<<<(NSEQ * Esz + 255) / 256, 256, 0, stream>>>(pooled, proj_W, proj_b, zbuf);
  k_norm<<<NSEQ, 256, 0, stream>>>(zbuf, zn, znT);

  // similarity via WMMA: sim = (zn @ znT) / TAU
  dim3 grid_sim(NSEQ / BN, NSEQ / BM);  // 16 x 16
  k_gemm_bf16<<<grid_sim, 128, 0, stream>>>(zn, znT, sim, NSEQ, NSEQ, Esz, TAUI);

  // loss
  k_loss_rows<<<NSEQ, 256, 0, stream>>>(sim, labels, li_arr, hp_arr);
  k_final<<<1, 256, 0, stream>>>(li_arr, hp_arr, out);
}
